// Informer_53042846105793
// MI455X (gfx1250) — matrix operand order repaired and checked by reading, not compiled, not run on hardware
//
#include <hip/hip_runtime.h>
#include <hip/hip_bf16.h>
#include <math.h>

// ---------------------------------------------------------------------------
// Informer encoder forward for MI455X (gfx1250, wave32, WMMA f16 16x16x32).
// Dense GEMMs (QKV/O projections, FFN) run on v_wmma_f32_16x16x32_f16 with
// pre-swizzled f16 fragments; softmax/top-k/LayerNorm stay fp32 VALU.
// Wave tile 32x64: 8 WMMA per K-step on 6 fragment loads (~22 FLOP/B).
// ---------------------------------------------------------------------------

#define BATCH   32
#define SEQL    1024
#define CIN     21
#define DMODEL  128
#define NHEADS  8
#define HEADD   16
#define DFF     512
#define NLAYERS 3
#define TOPU    35
#define SAMPK   35

typedef _Float16 v16h __attribute__((ext_vector_type(16)));
typedef float    v8f  __attribute__((ext_vector_type(8)));

__device__ __forceinline__ float gelu_exact(float x) {
    return 0.5f * x * (1.0f + erff(x * 0.70710678118654752440f));
}

// ---------------------------------------------------------------------------
// Threefry2x32: reproduce IDX_SAMPLE = jax.random.randint(key(42),(1024,35),0,1024)
// span=1024 divides 2^16 so the jax randint path reduces to bits & 1023.
// ---------------------------------------------------------------------------
__device__ __forceinline__ unsigned rotl32(unsigned x, int r) {
    return (x << r) | (x >> (32 - r));
}

__global__ void k_threefry_idx(int* __restrict__ idx) {
    int i = blockIdx.x * blockDim.x + threadIdx.x;
    const int half = (SEQL * SAMPK) / 2;   // 17920
    if (i >= half) return;
    unsigned x0 = (unsigned)i, x1 = (unsigned)(i + half);
    unsigned ks[3] = {0u, 42u, 0u ^ 42u ^ 0x1BD11BDAu};
    x0 += ks[0]; x1 += ks[1];
    const int rot[2][4] = {{13, 15, 26, 6}, {17, 29, 16, 24}};
    #pragma unroll
    for (int g = 0; g < 5; ++g) {
        const int* r = rot[g & 1];
        #pragma unroll
        for (int j = 0; j < 4; ++j) { x0 += x1; x1 = rotl32(x1, r[j]); x1 ^= x0; }
        x0 += ks[(g + 1) % 3];
        x1 += ks[(g + 2) % 3] + (unsigned)(g + 1);
    }
    idx[i]        = (int)(x0 & 1023u);
    idx[i + half] = (int)(x1 & 1023u);
}

// ---------------------------------------------------------------------------
// Embedding: circular conv (k=3) over C_IN=21 channels + sinusoidal PE.
// ---------------------------------------------------------------------------
__global__ void k_embed(const float* __restrict__ x, const float* __restrict__ w,
                        float* __restrict__ out) {
    int t = blockIdx.x * blockDim.x + threadIdx.x;
    if (t >= BATCH * SEQL * DMODEL) return;
    int d = t & (DMODEL - 1);
    int l = (t >> 7) & (SEQL - 1);
    int b = t >> 17;
    float s = 0.f;
    #pragma unroll
    for (int k = 0; k < 3; ++k) {
        int ls = (l + k + SEQL - 1) & (SEQL - 1);
        const float* xr = x + ((size_t)(b * SEQL + ls)) * CIN;
        const float* wr = w + (size_t)(k * CIN) * DMODEL + d;
        #pragma unroll
        for (int c = 0; c < CIN; ++c) s += xr[c] * wr[c * DMODEL];
    }
    int i = d >> 1;
    float div = expf(-(float)(2 * i) * (9.210340371976184f / (float)DMODEL)); // ln(1e4)
    float ang = (float)l * div;
    s += (d & 1) ? cosf(ang) : sinf(ang);
    out[t] = s;
}

// ---------------------------------------------------------------------------
// Fragment packing (contiguous k order, k = 16*(lane/16) + j, same for A and B).
// A (MxK f32, row major) -> per-fragment [(mt*KT+kt)*32 + lane] v16h.
//   lane<16: halves = K(0..15); lane>=16: halves = K(16..31).
// ---------------------------------------------------------------------------
__global__ void k_pack_a(const float* __restrict__ A, _Float16* __restrict__ P,
                         int M, int K) {
    int KT = K >> 5;
    int total = (M >> 4) * KT * 32;
    int t = blockIdx.x * blockDim.x + threadIdx.x;
    if (t >= total) return;
    int lane = t & 31, frag = t >> 5;
    int kt = frag % KT, mt = frag / KT;
    int row = mt * 16 + (lane & 15);
    int hi8 = ((lane >> 4) & 1) * 8;
    const float* src = A + (size_t)row * K + kt * 32;
    v16h o;
    #pragma unroll
    for (int j = 0; j < 16; ++j) {
        int k = 2 * hi8 + j;
        o[j] = (_Float16)src[k];
    }
    ((v16h*)P)[t] = o;
}

// B (KxN f32, row major) -> per-fragment [(nt*KT+kt)*32 + lane] v16h.
//   col = lane&15; k = (lane>=16?16:0) + j  (within the 32-K fragment).
__global__ void k_pack_b(const float* __restrict__ B, _Float16* __restrict__ P,
                         int K, int N) {
    int KT = K >> 5;
    int total = (N >> 4) * KT * 32;
    int t = blockIdx.x * blockDim.x + threadIdx.x;
    if (t >= total) return;
    int lane = t & 31, frag = t >> 5;
    int kt = frag % KT, nt = frag / KT;
    int col = nt * 16 + (lane & 15);
    int kbase = kt * 32 + ((lane >> 4) & 1) * 16;
    v16h o;
    #pragma unroll
    for (int j = 0; j < 16; ++j) o[j] = (_Float16)B[(size_t)(kbase + j) * N + col];
    ((v16h*)P)[t] = o;
}

// ---------------------------------------------------------------------------
// WMMA GEMM: C(MxN) = act( A(MxK) * B(KxN) + bias ).  mode: 0=bias, 1=bias+GELU.
// Wave tile 32x64 (2 m-fragments x 4 n-fragments), 8 waves / block.
// ---------------------------------------------------------------------------
__global__ void k_gemm(const _Float16* __restrict__ PA, const _Float16* __restrict__ PB,
                       const float* __restrict__ bias, float* __restrict__ C,
                       int M, int N, int K, int mode) {
    int lane = threadIdx.x & 31;
    int wv   = threadIdx.x >> 5;
    int W = blockIdx.x * 8 + wv;
    int mTiles = M >> 5, nBlks = N >> 6, KT = K >> 5;
    if (W >= mTiles * nBlks) return;
    int mt = W % mTiles, nb = W / mTiles;

    v8f acc[2][4];
    #pragma unroll
    for (int m = 0; m < 2; ++m)
        #pragma unroll
        for (int n = 0; n < 4; ++n)
            acc[m][n] = (v8f){0.f, 0.f, 0.f, 0.f, 0.f, 0.f, 0.f, 0.f};

    const v16h* Ap0 = (const v16h*)PA + (size_t)(mt * 2) * KT * 32 + lane;
    const v16h* Ap1 = Ap0 + (size_t)KT * 32;
    const v16h* Bp  = (const v16h*)PB;
    int nt0 = nb * 4;
    for (int kt = 0; kt < KT; ++kt) {
        // Prefetch next K-step's A fragments (streamed operand) into cache.
        __builtin_prefetch((const void*)(Ap0 + (size_t)(kt + 1) * 32), 0, 1);
        __builtin_prefetch((const void*)(Ap1 + (size_t)(kt + 1) * 32), 0, 1);
        v16h a0 = Ap0[(size_t)kt * 32];
        v16h a1 = Ap1[(size_t)kt * 32];
        #pragma unroll
        for (int n = 0; n < 4; ++n) {
            v16h b = Bp[((size_t)(nt0 + n) * KT + kt) * 32 + lane];
            acc[0][n] = __builtin_amdgcn_wmma_f32_16x16x32_f16(
                false, a0, false, b, (short)0, acc[0][n], false, false);
            acc[1][n] = __builtin_amdgcn_wmma_f32_16x16x32_f16(
                false, a1, false, b, (short)0, acc[1][n], false, false);
        }
    }
    int hi   = (lane >> 4) & 1;
    int colb = nb * 64 + (lane & 15);
    #pragma unroll
    for (int m = 0; m < 2; ++m) {
        int row = (mt * 2 + m) * 16 + hi * 8;
        #pragma unroll
        for (int n = 0; n < 4; ++n) {
            int col = colb + n * 16;
            float bs = bias ? bias[col] : 0.f;
            #pragma unroll
            for (int v = 0; v < 8; ++v) {
                float x = acc[m][n][v] + bs;
                if (mode == 1) x = gelu_exact(x);
                C[(size_t)(row + v) * N + col] = x;
            }
        }
    }
}

// ---------------------------------------------------------------------------
// ProbSparse attention (fp32).
// ---------------------------------------------------------------------------
__global__ void k_mscore(const float* __restrict__ Q, const float* __restrict__ K,
                         const int* __restrict__ idx, float* __restrict__ M) {
    int t = blockIdx.x * blockDim.x + threadIdx.x;
    if (t >= BATCH * NHEADS * SEQL) return;
    int l = t & (SEQL - 1);
    int h = (t >> 10) & (NHEADS - 1);
    int b = t >> 13;
    const float* q = Q + ((size_t)(b * SEQL + l)) * DMODEL + h * HEADD;
    float qr[HEADD];
    #pragma unroll
    for (int d = 0; d < HEADD; ++d) qr[d] = q[d];
    float mx = -INFINITY, sm = 0.f;
    for (int s = 0; s < SAMPK; ++s) {
        int kl = idx[l * SAMPK + s];
        const float* kr = K + ((size_t)(b * SEQL + kl)) * DMODEL + h * HEADD;
        float acc = 0.f;
        #pragma unroll
        for (int d = 0; d < HEADD; ++d) acc += qr[d] * kr[d];
        mx = fmaxf(mx, acc); sm += acc;
    }
    M[(size_t)(b * NHEADS + h) * SEQL + l] = mx - sm * (1.0f / (float)SEQL);
}

// Top-U indices per (b,h); ties -> smaller index (matches lax.top_k).
__global__ void k_topu(const float* __restrict__ M, int* __restrict__ topidx) {
    int bh = blockIdx.x, t = threadIdx.x;
    __shared__ float vals[SEQL];
    __shared__ float rv[256];
    __shared__ int   ri[256];
    for (int i = t; i < SEQL; i += 256) vals[i] = M[(size_t)bh * SEQL + i];
    __syncthreads();
    for (int u = 0; u < TOPU; ++u) {
        float bv = -INFINITY; int bi = SEQL;
        for (int i = t; i < SEQL; i += 256) {
            float v = vals[i];
            if (v > bv || (v == bv && i < bi)) { bv = v; bi = i; }
        }
        rv[t] = bv; ri[t] = bi; __syncthreads();
        for (int s = 128; s > 0; s >>= 1) {
            if (t < s) {
                if (rv[t + s] > rv[t] || (rv[t + s] == rv[t] && ri[t + s] < ri[t])) {
                    rv[t] = rv[t + s]; ri[t] = ri[t + s];
                }
            }
            __syncthreads();
        }
        if (t == 0) { topidx[bh * TOPU + u] = ri[0]; vals[ri[0]] = -INFINITY; }
        __syncthreads();
    }
}

// ctx[b,l,h,:] = mean_l V[b,l,h,:]
__global__ void k_ctx_mean(const float* __restrict__ V, float* __restrict__ CTX) {
    int bh = blockIdx.x, t = threadIdx.x;
    int b = bh >> 3, h = bh & 7;
    __shared__ float red[256 * HEADD];
    __shared__ float vm[HEADD];
    float p[HEADD];
    #pragma unroll
    for (int d = 0; d < HEADD; ++d) p[d] = 0.f;
    for (int l = t; l < SEQL; l += 256) {
        const float* vr = V + ((size_t)(b * SEQL + l)) * DMODEL + h * HEADD;
        #pragma unroll
        for (int d = 0; d < HEADD; ++d) p[d] += vr[d];
    }
    #pragma unroll
    for (int d = 0; d < HEADD; ++d) red[t * HEADD + d] = p[d];
    __syncthreads();
    for (int s = 128; s > 0; s >>= 1) {
        if (t < s) {
            #pragma unroll
            for (int d = 0; d < HEADD; ++d) red[t * HEADD + d] += red[(t + s) * HEADD + d];
        }
        __syncthreads();
    }
    if (t < HEADD) vm[t] = red[t] * (1.0f / (float)SEQL);
    __syncthreads();
    for (int l = t; l < SEQL; l += 256) {
        float* cr = CTX + ((size_t)(b * SEQL + l)) * DMODEL + h * HEADD;
        #pragma unroll
        for (int d = 0; d < HEADD; ++d) cr[d] = vm[d];
    }
}

// Full softmax attention for the TOPU selected queries; scatter into CTX.
__global__ void k_attn_top(const float* __restrict__ Q, const float* __restrict__ K,
                           const float* __restrict__ V, const int* __restrict__ topidx,
                           float* __restrict__ CTX) {
    int bh = blockIdx.x, t = threadIdx.x;
    int b = bh >> 3, h = bh & 7;
    __shared__ float p[SEQL];
    __shared__ float qs[HEADD];
    __shared__ float red[256];
    __shared__ float part[256];
    __shared__ int   sIdx;
    __shared__ float sMax, sSum;
    const float scale = 0.25f; // 1/sqrt(16)
    for (int u = 0; u < TOPU; ++u) {
        if (t == 0) sIdx = topidx[bh * TOPU + u];
        __syncthreads();
        int qrow = sIdx;
        if (t < HEADD) qs[t] = Q[((size_t)(b * SEQL + qrow)) * DMODEL + h * HEADD + t];
        __syncthreads();
        float lmax = -INFINITY;
        for (int l = t; l < SEQL; l += 256) {
            const float* kr = K + ((size_t)(b * SEQL + l)) * DMODEL + h * HEADD;
            float s = 0.f;
            #pragma unroll
            for (int d = 0; d < HEADD; ++d) s += qs[d] * kr[d];
            s *= scale;
            p[l] = s;
            lmax = fmaxf(lmax, s);
        }
        red[t] = lmax; __syncthreads();
        for (int s = 128; s > 0; s >>= 1) { if (t < s) red[t] = fmaxf(red[t], red[t + s]); __syncthreads(); }
        if (t == 0) sMax = red[0];
        __syncthreads();
        float lsum = 0.f;
        for (int l = t; l < SEQL; l += 256) { float e = expf(p[l] - sMax); p[l] = e; lsum += e; }
        red[t] = lsum; __syncthreads();
        for (int s = 128; s > 0; s >>= 1) { if (t < s) red[t] += red[t + s]; __syncthreads(); }
        if (t == 0) sSum = red[0];
        __syncthreads();
        int d = t & 15, ch = t >> 4;
        float a = 0.f;
        int l0 = ch * 64;
        for (int l = l0; l < l0 + 64; ++l)
            a += p[l] * V[((size_t)(b * SEQL + l)) * DMODEL + h * HEADD + d];
        part[t] = a; __syncthreads();
        if (t < HEADD) {
            float o = 0.f;
            #pragma unroll
            for (int c = 0; c < 16; ++c) o += part[c * 16 + t];
            CTX[((size_t)(b * SEQL + qrow)) * DMODEL + h * HEADD + t] = o / sSum;
        }
        __syncthreads();
    }
}

// ---------------------------------------------------------------------------
// LayerNorm(x [+ y]) * g + b   (one 128-thread block per row; in-place safe)
// ---------------------------------------------------------------------------
__global__ void k_ln(const float* __restrict__ x, const float* __restrict__ y,
                     const float* __restrict__ g, const float* __restrict__ bb,
                     float* __restrict__ out) {
    int row = blockIdx.x, t = threadIdx.x;
    __shared__ float red[DMODEL];
    size_t base = (size_t)row * DMODEL + t;
    float v = x[base] + (y ? y[base] : 0.f);
    red[t] = v; __syncthreads();
    for (int s = 64; s > 0; s >>= 1) { if (t < s) red[t] += red[t + s]; __syncthreads(); }
    float mu = red[0] * (1.0f / (float)DMODEL);
    __syncthreads();
    float d = v - mu;
    red[t] = d * d; __syncthreads();
    for (int s = 64; s > 0; s >>= 1) { if (t < s) red[t] += red[t + s]; __syncthreads(); }
    float var = red[0] * (1.0f / (float)DMODEL);
    out[base] = d * rsqrtf(var + 1e-5f) * g[t] + bb[t];
}

__global__ void k_gelu_mask(float* __restrict__ x, const float* __restrict__ mark) {
    int t = blockIdx.x * blockDim.x + threadIdx.x;
    if (t >= BATCH * SEQL * DMODEL) return;
    int bl = t >> 7; // (b*SEQL + l)
    x[t] = gelu_exact(x[t]) * mark[bl];
}

// out[b,c] = sum_i act[b,i]*W[i,c] + bias[c]   (i over L*D = 131072)
__global__ void k_proj(const float* __restrict__ act, const float* __restrict__ W,
                       const float* __restrict__ bias, float* __restrict__ out) {
    int blk = blockIdx.x, t = threadIdx.x;
    int b = blk / 10, c = blk % 10;
    __shared__ float red[256];
    float s = 0.f;
    const size_t NTOT = (size_t)SEQL * DMODEL;
    for (size_t i = t; i < NTOT; i += 256) s += act[(size_t)b * NTOT + i] * W[i * 10 + c];
    red[t] = s; __syncthreads();
    for (int st = 128; st > 0; st >>= 1) { if (t < st) red[t] += red[t + st]; __syncthreads(); }
    if (t == 0) out[b * 10 + c] = red[0] + bias[c];
}

// ---------------------------------------------------------------------------
// Host orchestration
// ---------------------------------------------------------------------------
struct DenseP { const float* w; const float* b; };
struct LayerP {
    DenseP q, k, v, o, c1, c2;
    const float *ln1g, *ln1b, *ln2g, *ln2b;
};

extern "C" void kernel_launch(void* const* d_in, const int* in_sizes, int n_in,
                              void* d_out, int out_size, void* d_ws, size_t ws_size,
                              hipStream_t stream) {
    (void)n_in; (void)out_size; (void)ws_size;
    auto F = [&](int i) { return (const float*)d_in[i]; };

    // --- Detect flatten convention from in_sizes (deterministic, host-only) ---
    const int XENC_N = BATCH * SEQL * CIN;
    bool topX = (in_sizes[0] == XENC_N);                 // x_enc first?
    int p0 = topX ? 2 : 0;                               // index of conv_w
    bool sortedNested = (in_sizes[p0 + 1] == DMODEL);    // first attn leaf: b(128) vs w(16384)

    const float* conv_w = F(p0);
    LayerP Lp[NLAYERS];
    int ix = p0 + 1;
    for (int L = 0; L < NLAYERS; ++L) {
        LayerP& P = Lp[L];
        if (sortedNested) { // keys sorted: attn{k,o,q,v} each {b,w}; c1{b,w}; c2{b,w}; ln1{b,g}; ln2{b,g}
            P.k.b = F(ix++); P.k.w = F(ix++);
            P.o.b = F(ix++); P.o.w = F(ix++);
            P.q.b = F(ix++); P.q.w = F(ix++);
            P.v.b = F(ix++); P.v.w = F(ix++);
            P.c1.b = F(ix++); P.c1.w = F(ix++);
            P.c2.b = F(ix++); P.c2.w = F(ix++);
            P.ln1b = F(ix++); P.ln1g = F(ix++);
            P.ln2b = F(ix++); P.ln2g = F(ix++);
        } else {            // insertion order: q,k,v,o each {w,b}; c1{w,b}; c2{w,b}; ln1{g,b}; ln2{g,b}
            P.q.w = F(ix++); P.q.b = F(ix++);
            P.k.w = F(ix++); P.k.b = F(ix++);
            P.v.w = F(ix++); P.v.b = F(ix++);
            P.o.w = F(ix++); P.o.b = F(ix++);
            P.c1.w = F(ix++); P.c1.b = F(ix++);
            P.c2.w = F(ix++); P.c2.b = F(ix++);
            P.ln1g = F(ix++); P.ln1b = F(ix++);
            P.ln2g = F(ix++); P.ln2b = F(ix++);
        }
    }
    const float *normg, *normb, *projw, *projb;
    if (sortedNested) { normb = F(ix++); normg = F(ix++); projb = F(ix++); projw = F(ix++); }
    else              { normg = F(ix++); normb = F(ix++); projw = F(ix++); projb = F(ix++); }
    const float* x_enc  = topX ? F(0) : F(ix++);
    const float* x_mark = topX ? F(1) : F(ix++);

    // --- Workspace layout ---
    char* ws = (char*)d_ws;
    const size_t ROWS = (size_t)BATCH * SEQL;            // 32768
    size_t o_X    = 0;                                   // 16 MiB f32 activations
    size_t o_PACK = o_X    + ROWS * DMODEL * 4;          // 32 MiB packed-A f16 (max K=512)
    size_t o_BIG  = o_PACK + ROWS * DFF * 2;             // 64 MiB (Q/K/V/CTX or FFN hidden)
    size_t o_TMP  = o_BIG  + ROWS * DFF * 4;             // 16 MiB (gemm outputs)
    size_t o_WPK  = o_TMP  + ROWS * DMODEL * 4;          // packed weights (f16)
    size_t wpkHalvesPerLayer = 4 * (size_t)DMODEL * DMODEL + 2 * (size_t)DMODEL * DFF;
    size_t o_M    = o_WPK  + NLAYERS * wpkHalvesPerLayer * 2;
    size_t o_IDX  = o_M    + (size_t)BATCH * NHEADS * SEQL * 4;
    size_t o_TOP  = o_IDX  + (size_t)SEQL * SAMPK * 4;

    float*    X    = (float*)(ws + o_X);
    _Float16* PACK = (_Float16*)(ws + o_PACK);
    float*    Qb   = (float*)(ws + o_BIG);
    float*    Kb   = Qb + ROWS * DMODEL;
    float*    Vb   = Kb + ROWS * DMODEL;
    float*    CTX  = Vb + ROWS * DMODEL;
    float*    Hb   = (float*)(ws + o_BIG);               // FFN hidden reuses Q..CTX
    float*    TMP  = (float*)(ws + o_TMP);
    float*    Mb   = (float*)(ws + o_M);
    int*      IDX  = (int*)(ws + o_IDX);
    int*      TOP  = (int*)(ws + o_TOP);

    // Packed weight pointers per layer
    _Float16* WPK[NLAYERS][6];
    {
        _Float16* p = (_Float16*)(ws + o_WPK);
        for (int L = 0; L < NLAYERS; ++L) {
            WPK[L][0] = p; p += (size_t)DMODEL * DMODEL;  // q
            WPK[L][1] = p; p += (size_t)DMODEL * DMODEL;  // k
            WPK[L][2] = p; p += (size_t)DMODEL * DMODEL;  // v
            WPK[L][3] = p; p += (size_t)DMODEL * DMODEL;  // o
            WPK[L][4] = p; p += (size_t)DMODEL * DFF;     // c1
            WPK[L][5] = p; p += (size_t)DFF * DMODEL;     // c2
        }
    }

    auto packB = [&](const float* w, _Float16* pw, int K, int N) {
        int total = (N >> 4) * (K >> 5) * 32;
        k_pack_b<<<(total + 255) / 256, 256, 0, stream>>>(w, pw, K, N);
    };
    auto packA = [&](const float* a, int M, int K) {
        int total = (M >> 4) * (K >> 5) * 32;
        k_pack_a<<<(total + 255) / 256, 256, 0, stream>>>(a, PACK, M, K);
    };
    auto gemm = [&](const _Float16* pb, const float* bias, float* C,
                    int M, int N, int K, int mode) {
        int waves = (M >> 5) * (N >> 6);   // 32x64 wave tiles
        k_gemm<<<(waves + 7) / 8, 256, 0, stream>>>(PACK, pb, bias, C, M, N, K, mode);
    };

    // --- Constants + embedding ---
    k_threefry_idx<<<(SEQL * SAMPK / 2 + 255) / 256, 256, 0, stream>>>(IDX);
    {
        int total = BATCH * SEQL * DMODEL;
        k_embed<<<(total + 255) / 256, 256, 0, stream>>>(x_enc, conv_w, X);
    }
    // --- Pack all weights ---
    for (int L = 0; L < NLAYERS; ++L) {
        packB(Lp[L].q.w,  WPK[L][0], DMODEL, DMODEL);
        packB(Lp[L].k.w,  WPK[L][1], DMODEL, DMODEL);
        packB(Lp[L].v.w,  WPK[L][2], DMODEL, DMODEL);
        packB(Lp[L].o.w,  WPK[L][3], DMODEL, DMODEL);
        packB(Lp[L].c1.w, WPK[L][4], DMODEL, DFF);
        packB(Lp[L].c2.w, WPK[L][5], DFF, DMODEL);
    }

    // --- Encoder layers ---
    const int M = (int)ROWS;
    for (int L = 0; L < NLAYERS; ++L) {
        const LayerP& P = Lp[L];
        // Q,K,V projections (WMMA)
        packA(X, M, DMODEL);
        gemm(WPK[L][0], P.q.b, Qb, M, DMODEL, DMODEL, 0);
        gemm(WPK[L][1], P.k.b, Kb, M, DMODEL, DMODEL, 0);
        gemm(WPK[L][2], P.v.b, Vb, M, DMODEL, DMODEL, 0);
        // ProbSparse attention (fp32)
        {
            int total = BATCH * NHEADS * SEQL;
            k_mscore<<<(total + 255) / 256, 256, 0, stream>>>(Qb, Kb, IDX, Mb);
        }
        k_topu<<<BATCH * NHEADS, 256, 0, stream>>>(Mb, TOP);
        k_ctx_mean<<<BATCH * NHEADS, 256, 0, stream>>>(Vb, CTX);
        k_attn_top<<<BATCH * NHEADS, 256, 0, stream>>>(Qb, Kb, Vb, TOP, CTX);
        // Output projection + residual + LN1
        packA(CTX, M, DMODEL);
        gemm(WPK[L][3], P.o.b, TMP, M, DMODEL, DMODEL, 0);
        k_ln<<<M, DMODEL, 0, stream>>>(X, TMP, P.ln1g, P.ln1b, X);
        // FFN (WMMA, GELU fused in first gemm) + residual + LN2
        packA(X, M, DMODEL);
        gemm(WPK[L][4], P.c1.b, Hb, M, DFF, DMODEL, 1);
        packA(Hb, M, DFF);
        gemm(WPK[L][5], P.c2.b, TMP, M, DMODEL, DFF, 0);
        k_ln<<<M, DMODEL, 0, stream>>>(X, TMP, P.ln2g, P.ln2b, X);
    }

    // --- Final norm, GELU*mask, classifier head ---
    k_ln<<<M, DMODEL, 0, stream>>>(X, nullptr, normg, normb, X);
    {
        int total = BATCH * SEQL * DMODEL;
        k_gelu_mask<<<(total + 255) / 256, 256, 0, stream>>>(X, x_mark);
    }
    k_proj<<<BATCH * 10, 256, 0, stream>>>(X, projw, projb, (float*)d_out);
}
